// Uni3DMultimodal_74646531604889
// MI455X (gfx1250) — matrix operand order repaired and checked by reading, not compiled, not run on hardware
//
#include <hip/hip_runtime.h>
#include <hip/hip_bf16.h>
#include <math.h>

// ---------------------------------------------------------------------------
// Types for CDNA5 WMMA (wave32)
// ---------------------------------------------------------------------------
typedef _Float16 f16;
typedef __attribute__((ext_vector_type(4)))  _Float16 v4h;
typedef __attribute__((ext_vector_type(8)))  _Float16 v8h;
typedef __attribute__((ext_vector_type(16))) _Float16 v16h;
typedef __attribute__((ext_vector_type(8)))  float    v8f;

__device__ __forceinline__ v16h cat16(v8h lo, v8h hi) {
  v16h r;
#pragma unroll
  for (int i = 0; i < 8; ++i) { r[i] = lo[i]; r[i + 8] = hi[i]; }
  return r;
}

// Problem constants
constexpr int BATCH = 4;
constexpr int NH    = 16;
constexpr int HDIM  = 64;
constexpr int DMODEL= 1024;
constexpr int LL1 = 576, LL2 = 512, LL3 = 77;
constexpr int LTOT = LL1 + LL2 + LL3;      // 1165
constexpr float CLIPV = 10000.0f;

// ---------------------------------------------------------------------------
// 1) sanitize + LayerNorm -> residual (f32) and normalized f16 activations
// ---------------------------------------------------------------------------
template<bool SAN>
__global__ __launch_bounds__(256)
void ln_kernel(const float* __restrict__ x, const float* __restrict__ g,
               const float* __restrict__ b, float* __restrict__ xs_out,
               f16* __restrict__ xn_out, int ntok)
{
  int t = blockIdx.x;
  if (t >= ntok) return;
  const float* xr = x + (size_t)t * DMODEL;
  int tid = threadIdx.x;
  __shared__ float red[256];

  float v[4];
  float s = 0.f;
#pragma unroll
  for (int i = 0; i < 4; ++i) {
    float e = xr[tid + 256 * i];
    if (SAN) e = isfinite(e) ? e : 0.0f;   // nan_to_num(nan=0, +/-inf=0)
    v[i] = e; s += e;
  }
  red[tid] = s; __syncthreads();
  for (int off = 128; off > 0; off >>= 1) {
    if (tid < off) red[tid] += red[tid + off];
    __syncthreads();
  }
  float mean = red[0] * (1.0f / DMODEL);
  __syncthreads();
  float sq = 0.f;
#pragma unroll
  for (int i = 0; i < 4; ++i) { float d = v[i] - mean; sq += d * d; }
  red[tid] = sq; __syncthreads();
  for (int off = 128; off > 0; off >>= 1) {
    if (tid < off) red[tid] += red[tid + off];
    __syncthreads();
  }
  float inv = rsqrtf(red[0] * (1.0f / DMODEL) + 1e-6f);
#pragma unroll
  for (int i = 0; i < 4; ++i) {
    int c = tid + 256 * i;
    float o = (v[i] - mean) * inv * g[c] + b[c];
    xn_out[(size_t)t * DMODEL + c] = (f16)o;
    if (SAN) xs_out[(size_t)t * DMODEL + c] = v[i];
  }
}

// ---------------------------------------------------------------------------
// 2) f32 -> f16 weight conversion (grid-stride)
// ---------------------------------------------------------------------------
__global__ void conv_f16_kernel(const float* __restrict__ a, f16* __restrict__ o, int n)
{
  for (int i = blockIdx.x * blockDim.x + threadIdx.x; i < n; i += gridDim.x * blockDim.x)
    o[i] = (f16)a[i];
}

// ---------------------------------------------------------------------------
// 3) Tiled WMMA GEMM:  C[M,N] = A[M,K] * W[N,K]^T  (+epilogue)
//    128x128 block tile, 8 waves (2x4), each wave 64x32 (4x2 WMMA tiles),
//    K-step 32 with v_wmma_f32_16x16x32_f16.
//    W tile is staged with the CDNA5 async global->LDS DMA path
//    (global_load_async_to_lds_b128, ASYNCcnt), A tile synchronously
//    (needs zero-fill for M-boundary rows).
// ---------------------------------------------------------------------------
enum { EPI_QKV = 0, EPI_PROJ = 1, EPI_GELU = 2, EPI_MLP2 = 3 };

template<int EPI>
__global__ __launch_bounds__(256)
void gemm_wmma(const f16* __restrict__ A, const f16* __restrict__ W,
               const float* __restrict__ bias, const float* __restrict__ resid,
               float* __restrict__ outF, f16* __restrict__ outH,
               int M, int N, int K,
               int rowsPerBatch, int batchRowStride, int rowOffset)
{
  __shared__ f16 As[128][40];   // padded, 16B-aligned rows (80B stride)
  __shared__ f16 Bs[128][40];

  int tid  = threadIdx.x;
  int lane = tid & 31, wave = tid >> 5;
  int wm = wave & 1, wn = wave >> 1;          // 2 x 4 wave grid
  int m0 = blockIdx.y * 128, n0 = blockIdx.x * 128;

  v8f acc[4][2] = {};

  // cooperative-load coordinates: thread -> (row, 16-element K chunk)
  int lrow = tid >> 1;
  int lk   = (tid & 1) * 16;

  int  lm   = m0 + lrow;
  bool aval = lm < M;
  size_t arow = 0;
  if (aval) {
    int bb = lm / rowsPerBatch;
    int rr = lm - bb * rowsPerBatch;
    arow = (size_t)(bb * batchRowStride + rowOffset + rr);
  }
  const f16* aptr = A + arow * (size_t)K + lk;
  const f16* wptr = W + (size_t)(n0 + lrow) * K + lk;

  // LDS byte address of this thread's Bs destination (generic ptr truncates
  // to the wave-relative LDS offset per the flat-aperture rules).
  unsigned bsAddr = (unsigned)(uintptr_t)&Bs[lrow][lk];

  int nk = K >> 5;
  for (int kt = 0; kt < nk; ++kt) {
    __syncthreads();
    // ---- A tile: synchronous (zero-fill beyond M) ----
    uint4 a0 = make_uint4(0, 0, 0, 0), a1 = a0;
    if (aval) {
      const uint4* p = (const uint4*)(aptr + (size_t)kt * 32);
      a0 = p[0]; a1 = p[1];
    }
    *(uint4*)&As[lrow][lk]     = a0;
    *(uint4*)&As[lrow][lk + 8] = a1;
    // ---- W tile: async global->LDS DMA (32B per thread, two b128 ops;
    //      INST_OFFSET is added to both the global and LDS address) ----
    {
      unsigned long long ga = (unsigned long long)(uintptr_t)(wptr + (size_t)kt * 32);
      asm volatile("global_load_async_to_lds_b128 %0, %1, off"
                   :: "v"(bsAddr), "v"(ga) : "memory");
      asm volatile("global_load_async_to_lds_b128 %0, %1, off offset:16"
                   :: "v"(bsAddr), "v"(ga) : "memory");
    }
    if (kt + 1 < nk) {                 // global_prefetch_b8 for next A tile
      __builtin_prefetch(aptr + (size_t)(kt + 1) * 32, 0, 1);
    }
    asm volatile("s_wait_asynccnt 0x0" ::: "memory");
    __syncthreads();

    int half = lane >> 4, r = lane & 15;
    v16h af[4];
#pragma unroll
    for (int mt = 0; mt < 4; ++mt) {
      // A 16x32 layout: lane row = lane&15, VGPR0-3: K = half*8+0..7, VGPR4-7: K = 16+half*8+0..7
      const v8h* pa = (const v8h*)&As[wm * 64 + mt * 16 + r][half * 8];
      af[mt] = cat16(pa[0], pa[2]);
    }
    v16h bf[2];
#pragma unroll
    for (int nt = 0; nt < 2; ++nt) {
      // B 32x16 layout: lane col = lane&15, VGPR0-3: K = half*8+0..7, VGPR4-7: K = 16+half*8+0..7
      const v8h* pb = (const v8h*)&Bs[wn * 32 + nt * 16 + r][half * 8];
      bf[nt] = cat16(pb[0], pb[2]);
    }
#pragma unroll
    for (int mt = 0; mt < 4; ++mt)
#pragma unroll
      for (int nt = 0; nt < 2; ++nt)
        acc[mt][nt] = __builtin_amdgcn_wmma_f32_16x16x32_f16(
            false, af[mt], false, bf[nt], (short)0, acc[mt][nt], false, false);
  }

  // Epilogue. C/D layout: VGPR j -> row j (lanes 0-15) / j+8 (lanes 16-31), col lane&15.
  int half = lane >> 4, cl = lane & 15;
#pragma unroll
  for (int mt = 0; mt < 4; ++mt) {
#pragma unroll
    for (int nt = 0; nt < 2; ++nt) {
#pragma unroll
      for (int j = 0; j < 8; ++j) {
        int m = m0 + wm * 64 + mt * 16 + j + half * 8;
        int n = n0 + wn * 32 + nt * 16 + cl;
        if (m < M) {
          float v = acc[mt][nt][j];
          size_t oi = (size_t)m * N + n;
          if (EPI == EPI_QKV) {
            v = fminf(fmaxf(v, -CLIPV), CLIPV);
            outF[oi] = v;
          } else if (EPI == EPI_PROJ) {
            v += resid[oi] + bias[n];
            outF[oi] = v;
          } else if (EPI == EPI_GELU) {
            v += bias[n];
            float c3 = v * v * v;
            float gg = 0.5f * v * (1.0f + tanhf(0.7978845608f * (v + 0.044715f * c3)));
            outH[oi] = (f16)gg;
          } else { // EPI_MLP2
            v += resid[oi] + bias[n];
            v = fminf(fmaxf(v, -CLIPV), CLIPV);
            outF[oi] = v;
          }
        }
      }
    }
  }
}

// ---------------------------------------------------------------------------
// 4) QKV post-pass: RMSNorm(q,k) + RoPE, emit q/k/v as f16 in [B,H,L,64].
//    One wave32 per (b,h,token); lane handles pair d = lane (HD/2 = 32 pairs).
// ---------------------------------------------------------------------------
__global__ __launch_bounds__(256)
void qkv_post(const float* __restrict__ qkv, const float* __restrict__ qs,
              const float* __restrict__ ks, const float* __restrict__ pe,
              f16* __restrict__ qo, f16* __restrict__ ko, f16* __restrict__ vo,
              int Lx, int off)
{
  int lane = threadIdx.x & 31, wave = threadIdx.x >> 5;
  int idx = blockIdx.x * 8 + wave;
  int total = BATCH * NH * Lx;
  if (idx >= total) return;
  int b = idx / (NH * Lx);
  int rem = idx - b * NH * Lx;
  int h = rem / Lx;
  int l = rem - h * Lx;

  size_t base  = (((size_t)(b * Lx + l) * 3 + 0) * NH + h) * HDIM;  // q slot
  size_t swich = (size_t)NH * HDIM;                                  // -> k, v
  int gl = off + l;
  size_t obase = ((size_t)(b * NH + h) * LTOT + gl) * HDIM;
  size_t pb    = ((size_t)(b * LTOT + gl) * 32 + lane) * 4;
  float c00 = pe[pb], c01 = pe[pb + 1], c10 = pe[pb + 2], c11 = pe[pb + 3];

  // q: rmsnorm + rope
  {
    float e0 = qkv[base + 2 * lane], e1 = qkv[base + 2 * lane + 1];
    float s = e0 * e0 + e1 * e1;
    for (int o = 16; o; o >>= 1) s += __shfl_xor(s, o, 32);
    float r = rsqrtf(s * (1.0f / HDIM) + 1e-6f);
    float a0 = e0 * r * qs[2 * lane], a1 = e1 * r * qs[2 * lane + 1];
    qo[obase + 2 * lane]     = (f16)(c00 * a0 + c01 * a1);
    qo[obase + 2 * lane + 1] = (f16)(c10 * a0 + c11 * a1);
  }
  // k: rmsnorm + rope
  {
    float e0 = qkv[base + swich + 2 * lane], e1 = qkv[base + swich + 2 * lane + 1];
    float s = e0 * e0 + e1 * e1;
    for (int o = 16; o; o >>= 1) s += __shfl_xor(s, o, 32);
    float r = rsqrtf(s * (1.0f / HDIM) + 1e-6f);
    float a0 = e0 * r * ks[2 * lane], a1 = e1 * r * ks[2 * lane + 1];
    ko[obase + 2 * lane]     = (f16)(c00 * a0 + c01 * a1);
    ko[obase + 2 * lane + 1] = (f16)(c10 * a0 + c11 * a1);
  }
  // v: convert
  {
    vo[obase + 2 * lane]     = (f16)qkv[base + 2 * swich + 2 * lane];
    vo[obase + 2 * lane + 1] = (f16)qkv[base + 2 * swich + 2 * lane + 1];
  }
}

// ---------------------------------------------------------------------------
// 5) Flash attention with WMMA. Block = 4 waves, 64 queries; 32-key tiles.
//    scores = (Q Kt)/8 -> online softmax -> O += P V. Output f16 [B, L, D].
// ---------------------------------------------------------------------------
__global__ __launch_bounds__(128)
void flash_attn(const f16* __restrict__ Q, const f16* __restrict__ Kg,
                const f16* __restrict__ Vg, f16* __restrict__ O)
{
  __shared__ f16   Qs[64][72];       // 64 q rows x 64 hd (144B stride)
  __shared__ f16   Ks[32][72];       // 32 k rows x 64 hd
  __shared__ f16   Vst[64][40];      // transposed: [hd][key] (80B stride)
  __shared__ f16   Ps[4][16][40];    // per-wave probabilities 16x32
  __shared__ float Sraw[4][16][33];  // per-wave raw scores
  __shared__ float mrow[4][16], lrow[4][16], srow[4][16];

  int tid = threadIdx.x, lane = tid & 31, wave = tid >> 5;
  int bh = blockIdx.y;
  int b = bh >> 4, h = bh & 15;
  int q0 = blockIdx.x * 64;
  const size_t headbase = (size_t)bh * LTOT * HDIM;

  // Q tile load (zeros beyond L)
  {
    int row = tid >> 1, c0 = (tid & 1) * 32;
    int qg = q0 + row;
    uint4 z = make_uint4(0, 0, 0, 0), d0 = z, d1 = z, d2 = z, d3 = z;
    if (qg < LTOT) {
      const uint4* p = (const uint4*)(Q + headbase + (size_t)qg * HDIM + c0);
      d0 = p[0]; d1 = p[1]; d2 = p[2]; d3 = p[3];
    }
    uint4* o = (uint4*)&Qs[row][c0];
    o[0] = d0; o[1] = d1; o[2] = d2; o[3] = d3;
  }
  if (lane < 16) { mrow[wave][lane] = -INFINITY; lrow[wave][lane] = 0.f; }
  v8f oacc[4] = {};
  __syncthreads();

  int nkb = (LTOT + 31) >> 5;
  for (int kb = 0; kb < nkb; ++kb) {
    int kbase = kb << 5;
    // ---- K tile (row-major) ----
    {
      int row = tid >> 2, c0 = (tid & 3) * 16;
      int kg = kbase + row;
      uint4 z = make_uint4(0, 0, 0, 0), d0 = z, d1 = z;
      if (kg < LTOT) {
        const uint4* p = (const uint4*)(Kg + headbase + (size_t)kg * HDIM + c0);
        d0 = p[0]; d1 = p[1];
      }
      uint4* o = (uint4*)&Ks[row][c0];
      o[0] = d0; o[1] = d1;
    }
    // ---- V tile, transposed via per-thread 4x4 micro-transpose:
    //      4x 8B global loads, 4x ds_store_b64 ----
    {
      int kq = (tid & 7) * 4;    // key0 within tile
      int hq = (tid >> 3) * 4;   // hd0
      f16 vt[4][4];
#pragma unroll
      for (int i = 0; i < 4; ++i) {
        int kg2 = kbase + kq + i;
        v4h d = (v4h)(_Float16)0.f;
        if (kg2 < LTOT)
          d = *(const v4h*)(Vg + headbase + (size_t)kg2 * HDIM + hq);
#pragma unroll
        for (int j = 0; j < 4; ++j) vt[i][j] = d[j];
      }
#pragma unroll
      for (int j = 0; j < 4; ++j) {
        v4h o;
#pragma unroll
        for (int i = 0; i < 4; ++i) o[i] = vt[i][j];
        *(v4h*)&Vst[hq + j][kq] = o;
      }
    }
    __syncthreads();

    int half = lane >> 4, r = lane & 15;

    // scores: two 16x16 tiles, contraction over HD=64 (two K-steps of 32)
#pragma unroll
    for (int t = 0; t < 2; ++t) {
      v8f s = {};
#pragma unroll
      for (int kk = 0; kk < 2; ++kk) {
        const v8h* pa = (const v8h*)&Qs[wave * 16 + r][kk * 32 + half * 8];
        v16h aq = cat16(pa[0], pa[2]);
        const v8h* pb = (const v8h*)&Ks[t * 16 + r][kk * 32 + half * 8];
        v16h bk = cat16(pb[0], pb[2]);
        s = __builtin_amdgcn_wmma_f32_16x16x32_f16(false, aq, false, bk,
                                                   (short)0, s, false, false);
      }
#pragma unroll
      for (int j = 0; j < 8; ++j)
        Sraw[wave][j + half * 8][t * 16 + r] = s[j] * 0.125f;  // 1/sqrt(64)
    }

    // online softmax (lanes 0..15 each own one query row)
    if (lane < 16) {
      int qr = lane;
      float mo = mrow[wave][qr];
      int kvalid = LTOT - kbase; if (kvalid > 32) kvalid = 32;
      float mx = -INFINITY;
      for (int j = 0; j < kvalid; ++j) mx = fmaxf(mx, Sraw[wave][qr][j]);
      float mn = fmaxf(mo, mx);
      float scale, lacc = 0.f;
      if (mn == -INFINITY) {
        scale = 1.f;
        for (int j = 0; j < 32; ++j) Ps[wave][qr][j] = (f16)0.f;
      } else {
        scale = __expf(mo - mn);
        for (int j = 0; j < 32; ++j) {
          float p = 0.f;
          if (j < kvalid) { p = __expf(Sraw[wave][qr][j] - mn); lacc += p; }
          Ps[wave][qr][j] = (f16)p;
        }
      }
      mrow[wave][qr] = mn;
      lrow[wave][qr] = lrow[wave][qr] * scale + lacc;
      srow[wave][qr] = scale;
    }

    // rescale O accumulators (per-row factor from LDS)
#pragma unroll
    for (int d = 0; d < 4; ++d)
#pragma unroll
      for (int j = 0; j < 8; ++j)
        oacc[d][j] *= srow[wave][j + half * 8];

    // O += P(16x32) * V(32x64)
    {
      const v8h* pp = (const v8h*)&Ps[wave][r][half * 8];
      v16h ap = cat16(pp[0], pp[2]);
#pragma unroll
      for (int d = 0; d < 4; ++d) {
        const v8h* pv = (const v8h*)&Vst[d * 16 + r][half * 8];
        v16h bv = cat16(pv[0], pv[2]);
        oacc[d] = __builtin_amdgcn_wmma_f32_16x16x32_f16(false, ap, false, bv,
                                                         (short)0, oacc[d], false, false);
      }
    }
    __syncthreads();
  }

  // finalize: O /= l, clip, store f16 to attn [B, L, H*HD]
  int half = lane >> 4, r = lane & 15;
#pragma unroll
  for (int d = 0; d < 4; ++d) {
#pragma unroll
    for (int j = 0; j < 8; ++j) {
      int qr = j + half * 8;
      int qg = q0 + wave * 16 + qr;
      if (qg < LTOT) {
        float l = lrow[wave][qr];
        float v = (l > 0.f) ? oacc[d][j] / l : 0.f;
        v = fminf(fmaxf(v, -CLIPV), CLIPV);
        O[((size_t)(b * LTOT + qg)) * DMODEL + h * HDIM + d * 16 + r] = (f16)v;
      }
    }
  }
}

// ---------------------------------------------------------------------------
// Host orchestration
// ---------------------------------------------------------------------------
extern "C" void kernel_launch(void* const* d_in, const int* in_sizes, int n_in,
                              void* d_out, int out_size, void* d_ws, size_t ws_size,
                              hipStream_t stream)
{
  (void)in_sizes; (void)n_in; (void)out_size; (void)ws_size;

  const int Lx[3]  = { LL1, LL2, LL3 };
  const int off[3] = { 0, LL1, LL1 + LL2 };
  const int Ms[3]  = { BATCH * LL1, BATCH * LL2, BATCH * LL3 };

  const float* x[3] = { (const float*)d_in[0], (const float*)d_in[1], (const float*)d_in[2] };
  const float* pe   = (const float*)d_in[3];
  // per-stream params: [ln1_g, ln1_b, Wqkv, q_scale, k_scale, Wp, bp, ln2_g, ln2_b, W1, b1, W2, b2]
  const float* P[3][13];
  for (int s = 0; s < 3; ++s)
    for (int i = 0; i < 13; ++i)
      P[s][i] = (const float*)d_in[4 + 13 * s + i];

  // bump allocator over d_ws
  size_t pos = 0;
  auto alloc = [&](size_t bytes) -> void* {
    pos = (pos + 255) & ~(size_t)255;
    void* p = (char*)d_ws + pos;
    pos += bytes;
    return p;
  };

  float *xs[3], *qkv[3], *xnew[3];
  f16 *xn[3], *wqkv16[3], *wp16[3], *w116[3], *w216[3], *hbuf[3], *gbuf[3];
  for (int s = 0; s < 3; ++s) {
    xs[s]     = (float*)alloc((size_t)Ms[s] * DMODEL * 4);
    xn[s]     = (f16*)  alloc((size_t)Ms[s] * DMODEL * 2);
    wqkv16[s] = (f16*)  alloc((size_t)3 * DMODEL * DMODEL * 2);
    wp16[s]   = (f16*)  alloc((size_t)DMODEL * DMODEL * 2);
    w116[s]   = (f16*)  alloc((size_t)4 * DMODEL * DMODEL * 2);
    w216[s]   = (f16*)  alloc((size_t)4 * DMODEL * DMODEL * 2);
    qkv[s]    = (float*)alloc((size_t)Ms[s] * 3 * DMODEL * 4);
    xnew[s]   = (float*)alloc((size_t)Ms[s] * DMODEL * 4);
    hbuf[s]   = (f16*)  alloc((size_t)Ms[s] * DMODEL * 2);
    gbuf[s]   = (f16*)  alloc((size_t)Ms[s] * 4 * DMODEL * 2);
  }
  f16* qh   = (f16*)alloc((size_t)BATCH * NH * LTOT * HDIM * 2);
  f16* kh   = (f16*)alloc((size_t)BATCH * NH * LTOT * HDIM * 2);
  f16* vh   = (f16*)alloc((size_t)BATCH * NH * LTOT * HDIM * 2);
  f16* attn = (f16*)alloc((size_t)BATCH * LTOT * DMODEL * 2);

  float* outp[3];
  outp[0] = (float*)d_out;
  outp[1] = outp[0] + (size_t)Ms[0] * DMODEL;
  outp[2] = outp[1] + (size_t)Ms[1] * DMODEL;

  // 1) sanitize + LN1 + weight conversions
  for (int s = 0; s < 3; ++s) {
    ln_kernel<true><<<dim3(Ms[s]), 256, 0, stream>>>(x[s], P[s][0], P[s][1], xs[s], xn[s], Ms[s]);
    conv_f16_kernel<<<dim3(512), 256, 0, stream>>>(P[s][2],  wqkv16[s], 3 * DMODEL * DMODEL);
    conv_f16_kernel<<<dim3(512), 256, 0, stream>>>(P[s][5],  wp16[s],   DMODEL * DMODEL);
    conv_f16_kernel<<<dim3(512), 256, 0, stream>>>(P[s][9],  w116[s],   4 * DMODEL * DMODEL);
    conv_f16_kernel<<<dim3(512), 256, 0, stream>>>(P[s][11], w216[s],   4 * DMODEL * DMODEL);
  }

  // 2) QKV GEMM + 3) RMSNorm/RoPE per stream
  for (int s = 0; s < 3; ++s) {
    dim3 g((3 * DMODEL) / 128, (Ms[s] + 127) / 128);
    gemm_wmma<EPI_QKV><<<g, 256, 0, stream>>>(xn[s], wqkv16[s], nullptr, nullptr,
                                              qkv[s], nullptr, Ms[s], 3 * DMODEL, DMODEL,
                                              Ms[s], Ms[s], 0);
    int nwaves = BATCH * NH * Lx[s];
    qkv_post<<<dim3((nwaves + 7) / 8), 256, 0, stream>>>(qkv[s], P[s][3], P[s][4], pe,
                                                         qh, kh, vh, Lx[s], off[s]);
  }

  // 4) joint flash attention over all L=1165 tokens
  flash_attn<<<dim3((LTOT + 63) / 64, BATCH * NH), 128, 0, stream>>>(qh, kh, vh, attn);

  // 5) output projection (+residual), LN2, MLP
  for (int s = 0; s < 3; ++s) {
    dim3 gp(DMODEL / 128, (Ms[s] + 127) / 128);
    // attn rows for stream s of batch b live at (b*LTOT + off + l) -> row remap
    gemm_wmma<EPI_PROJ><<<gp, 256, 0, stream>>>((const f16*)attn, wp16[s], P[s][6], xs[s],
                                                xnew[s], nullptr, Ms[s], DMODEL, DMODEL,
                                                Lx[s], LTOT, off[s]);
    ln_kernel<false><<<dim3(Ms[s]), 256, 0, stream>>>(xnew[s], P[s][7], P[s][8],
                                                      nullptr, hbuf[s], Ms[s]);
    dim3 g1((4 * DMODEL) / 128, (Ms[s] + 127) / 128);
    gemm_wmma<EPI_GELU><<<g1, 256, 0, stream>>>(hbuf[s], w116[s], P[s][10], nullptr,
                                                nullptr, gbuf[s], Ms[s], 4 * DMODEL, DMODEL,
                                                Ms[s], Ms[s], 0);
    dim3 g2(DMODEL / 128, (Ms[s] + 127) / 128);
    gemm_wmma<EPI_MLP2><<<g2, 256, 0, stream>>>(gbuf[s], w216[s], P[s][12], xnew[s],
                                                outp[s], nullptr, Ms[s], DMODEL, 4 * DMODEL,
                                                Ms[s], Ms[s], 0);
  }
}
